// ProtoNet_MiniBLS_Ultimate_24653112279700
// MI455X (gfx1250) — matrix operand order repaired and checked by reading, not compiled, not run on hardware
//
#include <hip/hip_runtime.h>
#include <hip/hip_bf16.h>
#include <math.h>

typedef __attribute__((ext_vector_type(16))) __bf16 v16bf;
typedef __attribute__((ext_vector_type(4)))  __bf16 v4bf;
typedef __attribute__((ext_vector_type(8)))  float  v8f;
typedef __attribute__((ext_vector_type(4)))  float  v4f;

#define N_TOK 196
#define DDIM  768
#define PDIM  64
#define DFEAT 7232   // 1536 + 4160 + 1536
#define NSUP_ROWS 50   // B*n_sup
#define NQRY_ROWS 100  // B*n_qry

// ---------------- workspace layout (float offsets) ----------------
static const size_t OFF_GAP_S = 0;                       // 500*768
static const size_t OFF_BLS_S = 384000;                  // 500*768
static const size_t OFF_GAP_Q = 768000;                  // 1000*768
static const size_t OFF_BLS_Q = 1536000;                 // 1000*768
static const size_t OFF_X_S   = 2304000;                 // 500*196*64
static const size_t OFF_X_Q   = 8576000;                 // 1000*196*64
static const size_t OFF_CF_S  = 21120000;                // 500*2080
static const size_t OFF_CF_Q  = 22160000;                // 1000*2080
static const size_t OFF_Z_S   = 24240000;                // 50*7232
static const size_t OFF_Z_Q   = 24601600;                // 100*7232
static const size_t OFF_MU_S  = 25324800;                // 2*7232
static const size_t OFF_SD_S  = 25339264;
static const size_t OFF_MU_Q  = 25353728;
static const size_t OFF_SD_Q  = 25368192;

__global__ void zero_kernel(float* p, int n) {
    int i = blockIdx.x * 256 + threadIdx.x;
    if (i < n) p[i] = 0.f;
}

// ---------------------------------------------------------------------------
// Fused per-sample WMMA GEMM (bf16 in / f32 accumulate).
// IS_X=false : bls = sum_rows tanh(feat @ w_bls[:,128*nblk..] + b), grid(12, S)
//              (+ gap column sums folded into nblk==0 staging)
// IS_X=true  : x = feat @ cov_reducer (196x64), grid(2, S)
// Block 256 = 8 waves; each wave owns one 16-row M-tile x all N-tiles.
// LDS tiles are staged directly in WMMA fragment order:
//   [tile][lane][16 bf16 contiguous] -> fragment load = one 32B LDS read.
// ---------------------------------------------------------------------------
template<bool IS_X>
__global__ __launch_bounds__(256)
void fused_gemm_kernel(const float* __restrict__ feats,
                       const float* __restrict__ wbls,
                       const float* __restrict__ bbls,
                       const float* __restrict__ covred,
                       float* __restrict__ gap,
                       float* __restrict__ bls,
                       float* __restrict__ xout)
{
    constexpr int NT = IS_X ? 4 : 8;
    __shared__ __bf16 AsF[8 * 32 * 16];    // 8 M-tiles, fragment order
    __shared__ __bf16 BsF[NT * 32 * 16];   // NT N-tiles, fragment order
    __shared__ float  gcol[32];

    const int tid   = threadIdx.x;
    const int lane  = tid & 31;
    const int wt    = tid >> 5;            // wave id 0..7 = M-tile index
    const int lr    = lane & 15;
    const int mblk  = blockIdx.x & 1;
    const int nblk  = blockIdx.x >> 1;     // 0..5 (bls) / 0 (x)
    const int gs    = blockIdx.y;          // flat (l, sample)
    const int mbase = mblk * 128;
    const int nbase = nblk * 128;
    const float* feat = feats + (size_t)gs * N_TOK * DDIM;

    if (tid < 32) gcol[tid] = 0.f;
    __syncthreads();

    v8f acc[NT];
    #pragma unroll
    for (int i = 0; i < NT; ++i)
        #pragma unroll
        for (int j = 0; j < 8; ++j) acc[i][j] = 0.f;

    for (int kk = 0; kk < DDIM; kk += 32) {
        // ---- stage A: feat rows [mbase,mbase+128) x k [kk,kk+32) ----
        // float4 global loads; each maps to 4 consecutive fragment slots.
        #pragma unroll
        for (int it = 0; it < 4; ++it) {
            int e4 = tid + it * 256;       // 0..1023
            int r  = e4 >> 3;
            int c  = (e4 & 7) * 4;         // k within tile, multiple of 4
            int grow = mbase + r;
            v4f d;
            if (grow < N_TOK) d = *(const v4f*)(feat + (size_t)grow * DDIM + kk + c);
            else { d.x = 0.f; d.y = 0.f; d.z = 0.f; d.w = 0.f; }
            if (!IS_X && nblk == 0 && grow < N_TOK) {   // gap partial sums (f32)
                atomicAdd(&gcol[c + 0], d.x);
                atomicAdd(&gcol[c + 1], d.y);
                atomicAdd(&gcol[c + 2], d.z);
                atomicAdd(&gcol[c + 3], d.w);
            }
            // inverse of A fragment layout: lh = k[4], e = k[3:0]
            int tileA = r >> 4, m = r & 15;
            int lh = (c >> 4) & 1;
            int eb = c & 15;
            v4bf h;
            h.x = (__bf16)d.x; h.y = (__bf16)d.y; h.z = (__bf16)d.z; h.w = (__bf16)d.w;
            *(v4bf*)(&AsF[(tileA * 32 + (m | (lh << 4))) * 16 + eb]) = h;
        }
        // ---- stage B: fragment layout e = k&15, lane = col&15 | (k>>4)<<4 ----
        if (!IS_X) {
            #pragma unroll
            for (int it = 0; it < 4; ++it) {
                int e4 = tid + it * 256;   // 0..1023 (32 rows x 32 float4)
                int k  = e4 >> 5;
                int c  = (e4 & 31) * 4;
                v4f d = *(const v4f*)(wbls + (size_t)(kk + k) * DDIM + nbase + c);
                __bf16* bp = &BsF[((c >> 4) * 32 + (c & 15) + ((k >> 4) << 4)) * 16 + (k & 15)];
                bp[0]  = (__bf16)d.x;      // 4 consecutive cols -> 4 lanes, same e
                bp[16] = (__bf16)d.y;
                bp[32] = (__bf16)d.z;
                bp[48] = (__bf16)d.w;
            }
        } else {
            #pragma unroll
            for (int it = 0; it < 2; ++it) {
                int e4 = tid + it * 256;   // 0..511 (32 rows x 16 float4)
                int k  = e4 >> 4;
                int c  = (e4 & 15) * 4;
                v4f d = *(const v4f*)(covred + (size_t)(kk + k) * PDIM + c);
                __bf16* bp = &BsF[((c >> 4) * 32 + (c & 15) + ((k >> 4) << 4)) * 16 + (k & 15)];
                bp[0]  = (__bf16)d.x;
                bp[16] = (__bf16)d.y;
                bp[32] = (__bf16)d.z;
                bp[48] = (__bf16)d.w;
            }
        }
        __syncthreads();

        // ---- WMMA: A frag = one 32B LDS read; B frags likewise ----
        v16bf af = *(const v16bf*)(&AsF[(wt * 32 + lane) * 16]);
        #pragma unroll
        for (int nt = 0; nt < NT; ++nt) {
            v16bf bfrag = *(const v16bf*)(&BsF[(nt * 32 + lane) * 16]);
            acc[nt] = __builtin_amdgcn_wmma_f32_16x16x32_bf16(
                false, af, false, bfrag, (short)0, acc[nt], false, false);
        }

        // flush gap partials for this K-step
        if (!IS_X && nblk == 0 && tid < 32) {
            atomicAdd(&gap[(size_t)gs * DDIM + kk + tid], gcol[tid]);
            gcol[tid] = 0.f;
        }
        __syncthreads();
    }

    // ---- epilogue ----
    if (!IS_X) {
        #pragma unroll
        for (int nt = 0; nt < NT; ++nt) {
            int col = nbase + nt * 16 + lr;
            float bias = bbls[col];
            float s = 0.f;
            #pragma unroll
            for (int v = 0; v < 8; ++v) {
                int grow = mbase + wt * 16 + ((lane < 16) ? v : v + 8);
                if (grow < N_TOK) s += tanhf(acc[nt][v] + bias);
            }
            s += __shfl_xor(s, 16, 32);    // combine lane halves (rows 0-7 / 8-15)
            if (lane < 16) atomicAdd(&bls[(size_t)gs * DDIM + col], s);
        }
    } else {
        #pragma unroll
        for (int nt = 0; nt < NT; ++nt) {
            int p = nt * 16 + lr;
            #pragma unroll
            for (int v = 0; v < 8; ++v) {
                int grow = mbase + wt * 16 + ((lane < 16) ? v : v + 8);
                if (grow < N_TOK)
                    xout[(size_t)gs * N_TOK * PDIM + (size_t)grow * PDIM + p] = acc[nt][v];
            }
        }
    }
}

// ---------------------------------------------------------------------------
// Per-sample: center x, cov = x^T x / 195 + 1e-5 I, Newton-Schulz sqrt (3 it),
// triu extraction. All in LDS (dynamic).
// ---------------------------------------------------------------------------
__global__ __launch_bounds__(256)
void cov_sqrt_kernel(const float* __restrict__ x, float* __restrict__ cf)
{
    extern __shared__ float sm[];
    float* xs   = sm;           // 12544 floats (196x64), reused below
    float* Ybuf = sm + 12544;   // 4096
    float* Zb   = sm;           // aliases xs (dead after cov)
    float* Tb   = sm + 4096;
    float* Wb   = sm + 8192;
    __shared__ float red[256];
    const int tid = threadIdx.x;
    const int gs  = blockIdx.x;
    const float* xg = x + (size_t)gs * N_TOK * PDIM;

    for (int i = tid; i < N_TOK * PDIM; i += 256) xs[i] = xg[i];
    __syncthreads();
    if (tid < 64) {                       // column means (over 196 tokens)
        float m = 0.f;
        for (int n = 0; n < N_TOK; ++n) m += xs[n * PDIM + tid];
        Ybuf[tid] = m / (float)N_TOK;
    }
    __syncthreads();
    for (int i = tid; i < N_TOK * PDIM; i += 256) xs[i] -= Ybuf[i & 63];
    __syncthreads();
    for (int o = tid; o < 4096; o += 256) {   // covariance into Ybuf
        int p = o >> 6, q = o & 63;
        float s = 0.f;
        for (int n = 0; n < N_TOK; ++n) s += xs[n * PDIM + p] * xs[n * PDIM + q];
        Ybuf[o] = s / 195.f + ((p == q) ? 1e-5f : 0.f);
    }
    __syncthreads();
    float ss = 0.f;                            // Frobenius norm
    for (int o = tid; o < 4096; o += 256) ss += Ybuf[o] * Ybuf[o];
    red[tid] = ss; __syncthreads();
    for (int st = 128; st > 0; st >>= 1) { if (tid < st) red[tid] += red[tid + st]; __syncthreads(); }
    float denom = sqrtf(red[0]) + 1e-8f;
    __syncthreads();
    for (int o = tid; o < 4096; o += 256) {
        int p = o >> 6, q = o & 63;
        Ybuf[o] /= denom;
        Zb[o] = (p == q) ? 1.f : 0.f;
    }
    __syncthreads();
    float *Y = Ybuf, *Z = Zb, *T = Tb, *W = Wb;
    for (int it = 0; it < 3; ++it) {
        for (int o = tid; o < 4096; o += 256) {   // T = 0.5*(3I - Z@Y)
            int p = o >> 6, q = o & 63;
            float s = 0.f;
            for (int j = 0; j < 64; ++j) s += Z[p * 64 + j] * Y[j * 64 + q];
            T[o] = 0.5f * (((p == q) ? 3.f : 0.f) - s);
        }
        __syncthreads();
        for (int o = tid; o < 4096; o += 256) {   // W = Y@T  (new Y)
            int p = o >> 6, q = o & 63;
            float s = 0.f;
            for (int j = 0; j < 64; ++j) s += Y[p * 64 + j] * T[j * 64 + q];
            W[o] = s;
        }
        __syncthreads();
        for (int o = tid; o < 4096; o += 256) {   // Y(old buf) = T@Z (new Z)
            int p = o >> 6, q = o & 63;
            float s = 0.f;
            for (int j = 0; j < 64; ++j) s += T[p * 64 + j] * Z[j * 64 + q];
            Y[o] = s;
        }
        __syncthreads();
        float* t0 = Y; Y = W; W = Z; Z = t0;      // rotate buffers
    }
    float scale = sqrtf(denom);
    for (int k = tid; k < 2080; k += 256) {       // triu(64) extraction
        int i = 0, rem = k;
        while (rem >= PDIM - i) { rem -= PDIM - i; ++i; }
        int j = i + rem;
        cf[(size_t)gs * 2080 + k] = Y[i * 64 + j] * scale;
    }
}

// ---------------------------------------------------------------------------
// Fuse: view-mean -> tukey -> per-component l2norm -> weight -> concat
// Grid: x = rows (50 or 100), y = component 0..2.  M = per-L sample count.
// ---------------------------------------------------------------------------
__global__ __launch_bounds__(256)
void fuse_kernel(const float* __restrict__ gap, const float* __restrict__ cf,
                 const float* __restrict__ bls, float* __restrict__ z, int M)
{
    __shared__ float red[256];
    const int tid = threadIdx.x;
    const int row = blockIdx.x;
    const int comp = blockIdx.y;
    int width, off; float w;
    if (comp == 0)      { width = 1536; off = 0;    w = 1.0f; }
    else if (comp == 1) { width = 4160; off = 1536; w = 0.8f; }
    else                { width = 1536; off = 5696; w = 0.1f; }
    float ss = 0.f;
    for (int c = tid; c < width; c += 256) {
        float val = 0.f;
        if (comp == 1) {
            int l = c / 2080, u = c % 2080;
            for (int v = 0; v < 5; ++v)
                val += cf[((size_t)(l * M + row * 5 + v)) * 2080 + u];
            val *= 0.2f;
        } else {
            const float* src = (comp == 0) ? gap : bls;
            int l = c / 768, d = c % 768;
            for (int v = 0; v < 5; ++v)
                val += src[((size_t)(l * M + row * 5 + v)) * 768 + d];
            val *= (1.f / (5.f * 196.f));   // /196 row-mean + /5 view-mean
        }
        float sgn = (val > 0.f) ? 1.f : ((val < 0.f) ? -1.f : 0.f);
        float t = sgn * sqrtf(fabsf(val) + 1e-8f);
        z[(size_t)row * DFEAT + off + c] = t;
        ss += t * t;
    }
    red[tid] = ss; __syncthreads();
    for (int st = 128; st > 0; st >>= 1) { if (tid < st) red[tid] += red[tid + st]; __syncthreads(); }
    float scale = w / fmaxf(sqrtf(red[0]), 1e-12f);
    for (int c = tid; c < width; c += 256)
        z[(size_t)row * DFEAT + off + c] *= scale;
}

__global__ void stats_kernel(const float* __restrict__ zs, const float* __restrict__ zq,
                             float* mus, float* sds, float* muq, float* sdq)
{
    int col = blockIdx.x * 256 + threadIdx.x;
    int b = blockIdx.y;
    if (col >= DFEAT) return;
    float m = 0.f;
    for (int r = 0; r < 25; ++r) m += zs[(size_t)(b * 25 + r) * DFEAT + col];
    m /= 25.f;
    float v = 0.f;
    for (int r = 0; r < 25; ++r) { float d = zs[(size_t)(b * 25 + r) * DFEAT + col] - m; v += d * d; }
    mus[b * DFEAT + col] = m;
    sds[b * DFEAT + col] = sqrtf(v / 24.f) + 1e-8f;
    m = 0.f;
    for (int r = 0; r < 50; ++r) m += zq[(size_t)(b * 50 + r) * DFEAT + col];
    m /= 50.f;
    v = 0.f;
    for (int r = 0; r < 50; ++r) { float d = zq[(size_t)(b * 50 + r) * DFEAT + col] - m; v += d * d; }
    muq[b * DFEAT + col] = m;
    sdq[b * DFEAT + col] = sqrtf(v / 49.f) + 1e-8f;
}

// sup: v = z - mu_s ; qry: v = (z - mu_q)/sd_q*sd_s  (mu_s cancels) ; then l2norm
__global__ __launch_bounds__(256)
void rownorm_kernel(float* __restrict__ z, const float* __restrict__ mu,
                    const float* __restrict__ sdev, const float* __restrict__ sds,
                    int rows_per_b, int is_qry)
{
    __shared__ float red[256];
    const int tid = threadIdx.x;
    const int row = blockIdx.x;
    const int b = row / rows_per_b;
    float ss = 0.f;
    for (int c = tid; c < DFEAT; c += 256) {
        float v = z[(size_t)row * DFEAT + c] - mu[b * DFEAT + c];
        if (is_qry) v = v / sdev[b * DFEAT + c] * sds[b * DFEAT + c];
        z[(size_t)row * DFEAT + c] = v;
        ss += v * v;
    }
    red[tid] = ss; __syncthreads();
    for (int st = 128; st > 0; st >>= 1) { if (tid < st) red[tid] += red[tid + st]; __syncthreads(); }
    float scale = 1.f / fmaxf(sqrtf(red[0]), 1e-12f);
    for (int c = tid; c < DFEAT; c += 256) z[(size_t)row * DFEAT + c] *= scale;
}

// per-b: K = zs zs^T + lam I ; Gauss-Jordan solve alpha = K^-1 Y ; out = 20*sim@alpha
__global__ __launch_bounds__(256)
void solve_kernel(const float* __restrict__ zs, const float* __restrict__ zq,
                  const int* __restrict__ ysup, float* __restrict__ out)
{
    __shared__ float Km[625];
    __shared__ float A[125];
    __shared__ float fac[25];
    __shared__ float sim[1250];
    const int tid = threadIdx.x;
    const int b = blockIdx.x;
    const float* zsb = zs + (size_t)b * 25 * DFEAT;
    const float* zqb = zq + (size_t)b * 50 * DFEAT;
    const float lam = 0.1f * ((float)DFEAT / 500.f);
    for (int t = tid; t < 625; t += 256) {
        int i = t / 25, j = t % 25;
        float s = 0.f;
        for (int c = 0; c < DFEAT; ++c) s += zsb[(size_t)i * DFEAT + c] * zsb[(size_t)j * DFEAT + c];
        Km[t] = s + ((i == j) ? lam : 0.f);
    }
    for (int t = tid; t < 125; t += 256) {
        int i = t / 5, c = t % 5;
        A[t] = (ysup[b * 25 + i] == c) ? 1.f : 0.f;
    }
    __syncthreads();
    for (int p = 0; p < 25; ++p) {
        float inv = 1.f / Km[p * 25 + p];
        __syncthreads();
        if (tid < 25) Km[p * 25 + tid] *= inv;
        if (tid < 5)  A[p * 5 + tid] *= inv;
        __syncthreads();
        if (tid < 25 && tid != p) fac[tid] = Km[tid * 25 + p];
        __syncthreads();
        for (int t = tid; t < 625; t += 256) {
            int r = t / 25, j = t % 25;
            if (r != p) Km[t] -= fac[r] * Km[p * 25 + j];
        }
        for (int t = tid; t < 125; t += 256) {
            int r = t / 5, c = t % 5;
            if (r != p) A[t] -= fac[r] * A[p * 5 + c];
        }
        __syncthreads();
    }
    for (int t = tid; t < 1250; t += 256) {
        int q = t / 25, n = t % 25;
        float s = 0.f;
        for (int c = 0; c < DFEAT; ++c) s += zqb[(size_t)q * DFEAT + c] * zsb[(size_t)n * DFEAT + c];
        sim[t] = s;
    }
    __syncthreads();
    for (int t = tid; t < 250; t += 256) {
        int q = t / 5, c = t % 5;
        float s = 0.f;
        for (int n = 0; n < 25; ++n) s += sim[q * 25 + n] * A[n * 5 + c];
        out[(size_t)b * 250 + t] = 20.f * s;
    }
}

extern "C" void kernel_launch(void* const* d_in, const int* in_sizes, int n_in,
                              void* d_out, int out_size, void* d_ws, size_t ws_size,
                              hipStream_t stream) {
    (void)in_sizes; (void)n_in; (void)out_size; (void)ws_size;
    const float* feats_sup   = (const float*)d_in[0];
    const float* feats_qry   = (const float*)d_in[1];
    const float* cov_reducer = (const float*)d_in[2];
    const float* w_bls       = (const float*)d_in[3];
    const float* b_bls       = (const float*)d_in[4];
    const int*   y_sup       = (const int*)d_in[5];
    float* ws  = (float*)d_ws;
    float* out = (float*)d_out;

    // 1) zero atomic accumulators (gap_s, bls_s, gap_q, bls_q)
    zero_kernel<<<(2304000 + 255) / 256, 256, 0, stream>>>(ws, 2304000);

    // 2) fused WMMA GEMMs (support: 500 samples, query: 1000 samples)
    fused_gemm_kernel<false><<<dim3(12, 500), 256, 0, stream>>>(
        feats_sup, w_bls, b_bls, cov_reducer,
        ws + OFF_GAP_S, ws + OFF_BLS_S, ws + OFF_X_S);
    fused_gemm_kernel<true><<<dim3(2, 500), 256, 0, stream>>>(
        feats_sup, w_bls, b_bls, cov_reducer,
        ws + OFF_GAP_S, ws + OFF_BLS_S, ws + OFF_X_S);
    fused_gemm_kernel<false><<<dim3(12, 1000), 256, 0, stream>>>(
        feats_qry, w_bls, b_bls, cov_reducer,
        ws + OFF_GAP_Q, ws + OFF_BLS_Q, ws + OFF_X_Q);
    fused_gemm_kernel<true><<<dim3(2, 1000), 256, 0, stream>>>(
        feats_qry, w_bls, b_bls, cov_reducer,
        ws + OFF_GAP_Q, ws + OFF_BLS_Q, ws + OFF_X_Q);

    // 3) covariance + Newton-Schulz sqrt + triu
    const size_t covLds = (12544 + 4096) * sizeof(float);
    cov_sqrt_kernel<<<500, 256, covLds, stream>>>(ws + OFF_X_S, ws + OFF_CF_S);
    cov_sqrt_kernel<<<1000, 256, covLds, stream>>>(ws + OFF_X_Q, ws + OFF_CF_Q);

    // 4) fuse components into z_sup / z_qry
    fuse_kernel<<<dim3(NSUP_ROWS, 3), 256, 0, stream>>>(
        ws + OFF_GAP_S, ws + OFF_CF_S, ws + OFF_BLS_S, ws + OFF_Z_S, 250);
    fuse_kernel<<<dim3(NQRY_ROWS, 3), 256, 0, stream>>>(
        ws + OFF_GAP_Q, ws + OFF_CF_Q, ws + OFF_BLS_Q, ws + OFF_Z_Q, 500);

    // 5) stats + alignment + row l2norm
    stats_kernel<<<dim3((DFEAT + 255) / 256, 2), 256, 0, stream>>>(
        ws + OFF_Z_S, ws + OFF_Z_Q,
        ws + OFF_MU_S, ws + OFF_SD_S, ws + OFF_MU_Q, ws + OFF_SD_Q);
    rownorm_kernel<<<NSUP_ROWS, 256, 0, stream>>>(
        ws + OFF_Z_S, ws + OFF_MU_S, ws + OFF_SD_Q, ws + OFF_SD_S, 25, 0);
    rownorm_kernel<<<NQRY_ROWS, 256, 0, stream>>>(
        ws + OFF_Z_Q, ws + OFF_MU_Q, ws + OFF_SD_Q, ws + OFF_SD_S, 50, 1);

    // 6) ridge solve + scores
    solve_kernel<<<2, 256, 0, stream>>>(ws + OFF_Z_S, ws + OFF_Z_Q, y_sup, out);
}
